// SlideGraphArch_12953621365179
// MI455X (gfx1250) — matrix operand order repaired and checked by reading, not compiled, not run on hardware
//
#include <hip/hip_runtime.h>

// ---------------------------------------------------------------------------
// GIN forward for MI455X (gfx1250): bf16 WMMA GEMMs (f32 accumulate), fp32
// atomics for segment-sum + BN statistics, fused epilogues.
// Round 3: branch-hoisted GEMM epilogue with constant-offset stores; float4
// widening of the elementwise / scatter / tail kernels.
// ---------------------------------------------------------------------------

typedef __attribute__((ext_vector_type(16))) __bf16 v16bf;
typedef __attribute__((ext_vector_type(8)))  __bf16 v8bf;
typedef __attribute__((ext_vector_type(4)))  __bf16 v4bf;
typedef __attribute__((ext_vector_type(8)))  float  v8f;

__device__ __forceinline__ unsigned short f2bf_us(float f) {
  // round-to-nearest-even f32 -> bf16 (finite inputs)
  unsigned u = __builtin_bit_cast(unsigned, f);
  unsigned r = u + 0x7FFFu + ((u >> 16) & 1u);
  return (unsigned short)(r >> 16);
}
__device__ __forceinline__ void atomicAddF(float* p, float v) {
  __hip_atomic_fetch_add(p, v, __ATOMIC_RELAXED, __HIP_MEMORY_SCOPE_AGENT);
}

// ---------------------------------------------------------------------------
// Convert + shuffle the five 128x128 fp32 weight matrices into bf16 fragments
// laid out exactly as each WMMA lane consumes them:
//   dst[((m*8 + w)*4 + ks)*32*16 + l*16 + e] = W_m[k][col]
//   with h = l>>4, col = 16w + (l&15), k = 32ks + 16h + e
// so a lane's 16-element B fragment is 32 contiguous bytes.
// ---------------------------------------------------------------------------
__global__ __launch_bounds__(256)
void convert_weights_kernel(const float* __restrict__ w0, const float* __restrict__ w1,
                            const float* __restrict__ w2, const float* __restrict__ w3,
                            const float* __restrict__ w4, unsigned short* __restrict__ dst) {
  int i = blockIdx.x * 256 + threadIdx.x;
  if (i >= 5 * 16384) return;
  int e  = i & 15;
  int l  = (i >> 4) & 31;
  int ks = (i >> 9) & 3;
  int w  = (i >> 11) & 7;
  int m  = i >> 14;
  int h   = l >> 4;
  int col = w * 16 + (l & 15);
  int k   = ks * 32 + h * 16 + e;
  const float* src = (m == 0) ? w0 : (m == 1) ? w1 : (m == 2) ? w2 : (m == 3) ? w3 : w4;
  dst[i] = f2bf_us(src[k * 128 + col]);
}

// ---------------------------------------------------------------------------
// GEMM: Y[nrows x 128] = A[nrows x 128] @ W[128 x 128] + bias
//   - A fp32 in global, converted to bf16 while staging into LDS
//   - v_wmma_f32_16x16x32_bf16, K=128 in 4 steps
//   - per block: 64 rows; 8 waves each own a 16-column slice, 4 M-subtiles
//   - B fragments pre-shuffled in global: 2x b128 load per fragment
//   - optional per-column sum / sumsq atomics (BN stats), optional dup store
// ---------------------------------------------------------------------------
__global__ __launch_bounds__(256)
void gemm128_wmma_kernel(const float* __restrict__ A, const unsigned short* __restrict__ Wpk,
                         const float* __restrict__ bias, float* __restrict__ Y,
                         float* __restrict__ Ydup, float* __restrict__ stats, int nrows) {
  __shared__ __bf16 smA[64][136];  // 64x128 bf16, rows padded to 272B
  const int tid = threadIdx.x;
  const int Mbase = blockIdx.x * 64;
  const bool full = (Mbase + 64) <= nrows;

  // Stage A tile (fp32 -> bf16): 2048 float4 chunks, 8 per thread.
  for (int t = tid; t < 64 * 32; t += 256) {
    int r = t >> 5, q = t & 31;
    int gr = Mbase + r;
    float4 v = make_float4(0.f, 0.f, 0.f, 0.f);
    if (gr < nrows) v = ((const float4*)A)[(size_t)gr * 32 + q];
    v4bf b = { (__bf16)v.x, (__bf16)v.y, (__bf16)v.z, (__bf16)v.w };
    *(v4bf*)&smA[r][q * 4] = b;
  }
  __syncthreads();

  const int w = tid >> 5, l = tid & 31;
  const int nb = w * 16;        // column slice base for this wave
  const int col = l & 15;       // N index (and M index for A frags)
  const int hh = l >> 4;        // lane-half selector

  // B fragments: 32B contiguous per lane per kstep.
  v16bf bfrag[4];
#pragma unroll
  for (int ks = 0; ks < 4; ++ks) {
    const unsigned short* p = Wpk + (((w * 4 + ks) * 32) + l) * 16;
    v8bf b0 = *(const v8bf*)p;
    v8bf b1 = *(const v8bf*)(p + 8);
    bfrag[ks] = __builtin_shufflevector(b0, b1, 0, 1, 2, 3, 4, 5, 6, 7,
                                        8, 9, 10, 11, 12, 13, 14, 15);
  }

  const float bcol = bias[nb + col];
  const bool do_stats = (stats != nullptr);
  const bool do_dup = (Ydup != nullptr);
  float s1 = 0.0f, s2 = 0.0f;

#pragma unroll
  for (int sub = 0; sub < 4; ++sub) {
    v8f acc = {};
    const int arow = sub * 16 + col;  // A-matrix M = lane&15
#pragma unroll
    for (int ks = 0; ks < 4; ++ks) {
      // A frag: elements 0..15 -> K = 32ks + 16hh + e
      const int k0 = ks * 32 + hh * 16;
      v8bf lo = *(const v8bf*)&smA[arow][k0];
      v8bf hi = *(const v8bf*)&smA[arow][k0 + 8];
      v16bf a = __builtin_shufflevector(lo, hi, 0, 1, 2, 3, 4, 5, 6, 7,
                                        8, 9, 10, 11, 12, 13, 14, 15);
      acc = __builtin_amdgcn_wmma_f32_16x16x32_bf16(
          false, a, false, bfrag[ks], (short)0, acc, false, false);
    }
    // Epilogue: D layout lane l -> N = l&15, rows M = sub*16 + 8*hh + r
    const int mtop = sub * 16 + hh * 8;
    float vals[8];
#pragma unroll
    for (int r = 0; r < 8; ++r) vals[r] = acc[r] + bcol;

    const size_t base = (size_t)(Mbase + mtop) * 128 + nb + col;
    if (full) {
      float* __restrict__ yp = Y + base;
#pragma unroll
      for (int r = 0; r < 8; ++r) yp[r * 128] = vals[r];
      if (do_dup) {
        float* __restrict__ dp = Ydup + base;
#pragma unroll
        for (int r = 0; r < 8; ++r) dp[r * 128] = vals[r];
      }
      if (do_stats) {
#pragma unroll
        for (int r = 0; r < 8; ++r) { s1 += vals[r]; s2 += vals[r] * vals[r]; }
      }
    } else {
#pragma unroll
      for (int r = 0; r < 8; ++r) {
        if (Mbase + mtop + r < nrows) {
          Y[base + r * 128] = vals[r];
          if (do_dup) Ydup[base + r * 128] = vals[r];
          if (do_stats) { s1 += vals[r]; s2 += vals[r] * vals[r]; }
        }
      }
    }
  }

  if (do_stats) {  // columns split across lane pairs (l, l+16)
    s1 += __shfl_xor(s1, 16, 32);
    s2 += __shfl_xor(s2, 16, 32);
    if (hh == 0) {
      atomicAddF(&stats[nb + col], s1);
      atomicAddF(&stats[128 + nb + col], s2);
    }
  }
}

// ---------------------------------------------------------------------------
// BN finalize: stats(sum,sumsq) -> per-column (scale, shift)
// ---------------------------------------------------------------------------
__global__ void bn_finalize_kernel(const float* __restrict__ stats, const float* __restrict__ g,
                                   const float* __restrict__ be, float* __restrict__ ss,
                                   float invN) {
  int c = threadIdx.x;  // 128 threads
  float mean = stats[c] * invN;
  float var = stats[128 + c] * invN - mean * mean;
  float sc = g[c] * rsqrtf(var + 1e-5f);
  ss[c] = sc;
  ss[128 + c] = be[c] - mean * sc;
}

// ---------------------------------------------------------------------------
// Fused BN + ReLU, float4-wide (optional duplicate store for agg-buffer init)
// ---------------------------------------------------------------------------
__global__ __launch_bounds__(256)
void bnrelu_kernel(const float4* __restrict__ y, const float* __restrict__ ss,
                   float4* __restrict__ out, float4* __restrict__ dup, int n4) {
  int i = blockIdx.x * 256 + threadIdx.x;
  if (i >= n4) return;
  int c = (i * 4) & 127;
  float4 v = y[i];
  float4 r;
  r.x = fmaxf(fmaf(v.x, ss[c + 0], ss[128 + c + 0]), 0.0f);
  r.y = fmaxf(fmaf(v.y, ss[c + 1], ss[128 + c + 1]), 0.0f);
  r.z = fmaxf(fmaf(v.z, ss[c + 2], ss[128 + c + 2]), 0.0f);
  r.w = fmaxf(fmaf(v.w, ss[c + 3], ss[128 + c + 3]), 0.0f);
  out[i] = r;
  if (dup) dup[i] = r;
}

// ---------------------------------------------------------------------------
// Edge scatter: one wave per edge; agg[dst] += h[src]
// (lane reads 16B contiguous, 4 f32 atomics to consecutive dwords)
// ---------------------------------------------------------------------------
__global__ __launch_bounds__(256)
void scatter_edges_kernel(const long long* __restrict__ ei, const float* __restrict__ h,
                          float* __restrict__ agg, int E) {
  int we = (blockIdx.x * 256 + threadIdx.x) >> 5;
  int l = threadIdx.x & 31;
  if (we >= E) return;
  long long s = ei[we];
  long long d = ei[(size_t)E + we];
  const float* hs = h + (size_t)s * 128;
  float* ad = agg + (size_t)d * 128;
  const int f = l * 4;
  float4 v = *(const float4*)(hs + f);
  atomicAddF(&ad[f + 0], v.x);
  atomicAddF(&ad[f + 1], v.y);
  atomicAddF(&ad[f + 2], v.z);
  atomicAddF(&ad[f + 3], v.w);
}

// ---------------------------------------------------------------------------
// Tail GEMV: out[n, 0:4] = feat[n, :] @ tail_w + tail_b  (one wave per node)
// ---------------------------------------------------------------------------
__global__ __launch_bounds__(256)
void tail_kernel(const float* __restrict__ feat, const float* __restrict__ tw,
                 const float* __restrict__ tb, float* __restrict__ out, int nrows) {
  int n = (blockIdx.x * 256 + threadIdx.x) >> 5;
  int l = threadIdx.x & 31;
  if (n >= nrows) return;
  const int f = l * 4;
  float4 xv = *(const float4*)(feat + (size_t)n * 128 + f);
  float4 w0 = *(const float4*)(tw + (f + 0) * 4);
  float4 w1 = *(const float4*)(tw + (f + 1) * 4);
  float4 w2 = *(const float4*)(tw + (f + 2) * 4);
  float4 w3 = *(const float4*)(tw + (f + 3) * 4);
  float a0 = xv.x * w0.x + xv.y * w1.x + xv.z * w2.x + xv.w * w3.x;
  float a1 = xv.x * w0.y + xv.y * w1.y + xv.z * w2.y + xv.w * w3.y;
  float a2 = xv.x * w0.z + xv.y * w1.z + xv.z * w2.z + xv.w * w3.z;
  float a3 = xv.x * w0.w + xv.y * w1.w + xv.z * w2.w + xv.w * w3.w;
#pragma unroll
  for (int off = 16; off > 0; off >>= 1) {
    a0 += __shfl_xor(a0, off, 32);
    a1 += __shfl_xor(a1, off, 32);
    a2 += __shfl_xor(a2, off, 32);
    a3 += __shfl_xor(a3, off, 32);
  }
  if (l == 0) {
    float4 o = make_float4(a0 + tb[0], a1 + tb[1], a2 + tb[2], a3 + tb[3]);
    *(float4*)(out + (size_t)n * 4) = o;
  }
}

// ---------------------------------------------------------------------------
extern "C" void kernel_launch(void* const* d_in, const int* in_sizes, int n_in,
                              void* d_out, int out_size, void* d_ws, size_t ws_size,
                              hipStream_t stream) {
  const float*      x       = (const float*)d_in[0];
  const long long*  ei      = (const long long*)d_in[1];  // edge_index int64 (2,E)
  const float*      head_w  = (const float*)d_in[2];
  const float*      head_b  = (const float*)d_in[3];
  const float*      head_g  = (const float*)d_in[4];
  const float*      head_be = (const float*)d_in[5];
  const float*      conv0_w = (const float*)d_in[6];
  const float*      conv0_b = (const float*)d_in[7];
  const float*      conv0_g = (const float*)d_in[8];
  const float*      conv0_be= (const float*)d_in[9];
  const float*      lin0_w  = (const float*)d_in[10];
  const float*      lin0_b  = (const float*)d_in[11];
  const float*      conv1_w = (const float*)d_in[12];
  const float*      conv1_b = (const float*)d_in[13];
  const float*      conv1_g = (const float*)d_in[14];
  const float*      conv1_be= (const float*)d_in[15];
  const float*      lin1_w  = (const float*)d_in[16];
  const float*      lin1_b  = (const float*)d_in[17];
  const float*      tail_w  = (const float*)d_in[18];
  const float*      tail_b  = (const float*)d_in[19];

  const int N = in_sizes[0] / 128;   // 50000
  const int E = in_sizes[1] / 2;     // 600000
  const size_t NF = (size_t)N * 128;

  float* out  = (float*)d_out;       // [N,4] first
  float* feat = out + (size_t)N * 4; // [N,128] second; also used as scratch

  // Workspace layout
  float*          bufA  = (float*)d_ws;            // N*128
  float*          bufB  = bufA + NF;               // N*128
  float*          stats = bufB + NF;               // 3 * 256 (sum | sumsq)
  float*          ss    = stats + 3 * 256;         // 3 * 256 (scale | shift)
  unsigned short* wbf   = (unsigned short*)(ss + 3 * 256);  // 5 * 16384 bf16

  hipMemsetAsync(stats, 0, 3 * 256 * sizeof(float), stream);
  convert_weights_kernel<<<(5 * 16384 + 255) / 256, 256, 0, stream>>>(
      head_w, conv0_w, lin0_w, conv1_w, lin1_w, wbf);

  const int GB = (N + 63) / 64;
  const int n4 = (int)(NF / 4);
  const int G4 = (n4 + 255) / 256;
  const float invN = 1.0f / (float)N;

  // head: y = x @ head_w + b (stats0) -> feat region (scratch)
  gemm128_wmma_kernel<<<GB, 256, 0, stream>>>(x, wbf + 0 * 16384, head_b, feat,
                                              (float*)nullptr, stats + 0, N);
  bn_finalize_kernel<<<1, 128, 0, stream>>>(stats + 0, head_g, head_be, ss + 0, invN);
  // feat0 = bnrelu(y) -> bufA, and initialize agg0 = feat0 -> bufB
  bnrelu_kernel<<<G4, 256, 0, stream>>>((const float4*)feat, ss + 0,
                                        (float4*)bufA, (float4*)bufB, n4);
  // agg0 += scatter(feat0)
  scatter_edges_kernel<<<(E + 7) / 8, 256, 0, stream>>>(ei, bufA, bufB, E);

  // conv0: y = agg0 @ conv0_w + b (stats1) -> feat region
  gemm128_wmma_kernel<<<GB, 256, 0, stream>>>(bufB, wbf + 1 * 16384, conv0_b, feat,
                                              (float*)nullptr, stats + 256, N);
  bn_finalize_kernel<<<1, 128, 0, stream>>>(stats + 256, conv0_g, conv0_be, ss + 256, invN);
  bnrelu_kernel<<<G4, 256, 0, stream>>>((const float4*)feat, ss + 256,
                                        (float4*)bufA, (float4*)nullptr, n4);
  // lin0: z0 = h @ lin0_w + b -> bufB, duplicate store initializes agg1 -> feat region
  gemm128_wmma_kernel<<<GB, 256, 0, stream>>>(bufA, wbf + 2 * 16384, lin0_b, bufB,
                                              feat, (float*)nullptr, N);
  // agg1 += scatter(z0)
  scatter_edges_kernel<<<(E + 7) / 8, 256, 0, stream>>>(ei, bufB, feat, E);

  // conv1: y = agg1 @ conv1_w + b (stats2) -> bufA
  gemm128_wmma_kernel<<<GB, 256, 0, stream>>>(feat, wbf + 3 * 16384, conv1_b, bufA,
                                              (float*)nullptr, stats + 512, N);
  bn_finalize_kernel<<<1, 128, 0, stream>>>(stats + 512, conv1_g, conv1_be, ss + 512, invN);
  bnrelu_kernel<<<G4, 256, 0, stream>>>((const float4*)bufA, ss + 512,
                                        (float4*)bufB, (float4*)nullptr, n4);
  // lin1: feat = h @ lin1_w + b -> final feat output region
  gemm128_wmma_kernel<<<GB, 256, 0, stream>>>(bufB, wbf + 4 * 16384, lin1_b, feat,
                                              (float*)nullptr, (float*)nullptr, N);

  // tail: out = feat @ tail_w + tail_b
  tail_kernel<<<(N + 7) / 8, 256, 0, stream>>>(feat, tail_w, tail_b, out, N);
}
